// FNPGNNv8_E8_38019050504236
// MI455X (gfx1250) — matrix operand order repaired and checked by reading, not compiled, not run on hardware
//
#include <hip/hip_runtime.h>
#include <hip/hip_bf16.h>
#include <math.h>

typedef __attribute__((ext_vector_type(16))) _Float16 v16h;
typedef __attribute__((ext_vector_type(8)))  _Float16 v8h;
typedef __attribute__((ext_vector_type(8)))  float    v8f;

#define BATCH   512
#define NNODE   116
#define DDIM    128
#define NHEAD   4
#define HDPER   32
#define INDIM   125
#define KPOOL   9
#define HEADDIM 1280
#define BN      (BATCH*NNODE)
#define BN_RSQ  0.9999950000374997f   /* 1/sqrt(1+1e-5) */

__device__ __forceinline__ float elu_f(float v) { return v > 0.f ? v : (expf(v) - 1.f); }

// ---------------------------------------------------------------------------
// f32 -> f16 row-padded convert:  src[M,srcK] f32 -> dst[M,dstK] f16 (zero pad)
// ---------------------------------------------------------------------------
__global__ void cvt_pad_kernel(const float* __restrict__ src, _Float16* __restrict__ dst,
                               int M, int srcK, int dstK)
{
    size_t idx = (size_t)blockIdx.x * blockDim.x + threadIdx.x;
    size_t total = (size_t)M * dstK;
    if (idx >= total) return;
    int k = (int)(idx % dstK);
    size_t m = idx / dstK;
    dst[idx] = (k < srcK) ? (_Float16)src[m * (size_t)srcK + k] : (_Float16)0.f;
}

// ---------------------------------------------------------------------------
// weight transpose+convert: src[K,N] f32 -> dst[N,dstK] f16 (zero pad along K)
// ---------------------------------------------------------------------------
__global__ void cvt_transpose_kernel(const float* __restrict__ src, _Float16* __restrict__ dst,
                                     int K, int Nn, int dstK)
{
    size_t idx = (size_t)blockIdx.x * blockDim.x + threadIdx.x;
    size_t total = (size_t)Nn * dstK;
    if (idx >= total) return;
    int k = (int)(idx % dstK);
    size_t n = idx / dstK;
    dst[idx] = (k < K) ? (_Float16)src[(size_t)k * Nn + n] : (_Float16)0.f;
}

// ---------------------------------------------------------------------------
// WMMA GEMM:  C[M,N] = epi( A[M,Kp] @ B[Kp,N] )
//   Ah : f16 row-major [M,Kp]       (Kp multiple of 32, M multiple of 64)
//   Bt : f16 row-major [N,Kp]       (= B transposed; N multiple of 16)
// One wave owns a 64x16 C strip (4 sub-tiles): each B fragment feeds 4 WMMAs.
// A fragment  = two contiguous v8h loads  (K=e+16*hi, matching B)
// B fragment  = one contiguous v16h load  (column per lane, K=e+16*hi)
// Epilogue: +bias[n], optional ELU, optional v*scale[n]*smul+shift[n], +residual.
// ---------------------------------------------------------------------------
__device__ __forceinline__ v16h load_a_frag(const _Float16* __restrict__ arow, int k0, int hi)
{
    v8h a0 = *(const v8h*)(arow + k0 + 16 * hi);
    v8h a1 = *(const v8h*)(arow + k0 + 8 + 16 * hi);
    v16h av;
#pragma unroll
    for (int e = 0; e < 8; ++e) { av[e] = a0[e]; av[e + 8] = a1[e]; }
    return av;
}

__global__ void gemm_wmma_kernel(const _Float16* __restrict__ Ah, const _Float16* __restrict__ Bt,
                                 float* __restrict__ C,
                                 int M, int Nn, int Kp, int ldc,
                                 const float* __restrict__ bias, int act,
                                 const float* __restrict__ scale, float smul,
                                 const float* __restrict__ shift,
                                 const float* __restrict__ residual)
{
    const int lane   = threadIdx.x & 31;
    const int wave   = threadIdx.x >> 5;
    const int tilesN = Nn >> 4;
    const int tilesM = M >> 6;                       // 64-row macro tiles
    const int tile   = blockIdx.x * (blockDim.x >> 5) + wave;
    if (tile >= tilesM * tilesN) return;             // wave-uniform: EXEC stays all-ones
    const int tM = tile / tilesN;
    const int tN = tile % tilesN;
    const int hi = lane >> 4;
    const int lo = lane & 15;

    const _Float16* __restrict__ arow = Ah + ((size_t)(tM * 64 + lo)) * Kp;
    const _Float16* __restrict__ brow = Bt + ((size_t)(tN * 16 + lo)) * Kp;
    const size_t astep = (size_t)16 * Kp;            // next 16-row sub-tile

    v8f acc[4] = {};
    for (int k0 = 0; k0 < Kp; k0 += 32) {
        __builtin_prefetch(brow + k0 + 64, 0, 0);
        __builtin_prefetch(arow + k0 + 64, 0, 0);
        v16h bv = *(const v16h*)(brow + k0 + 16 * hi);
#pragma unroll
        for (int s = 0; s < 4; ++s) {
            v16h av = load_a_frag(arow + (size_t)s * astep, k0, hi);
            acc[s] = __builtin_amdgcn_wmma_f32_16x16x32_f16(false, av, false, bv,
                                                            (short)0, acc[s], false, false);
        }
    }

    const int n = tN * 16 + lo;
    float bn_s = scale ? (scale[n] * smul) : 0.f;
    float bn_t = shift ? shift[n] : 0.f;
    float bi   = bias ? bias[n] : 0.f;
#pragma unroll
    for (int s = 0; s < 4; ++s) {
#pragma unroll
        for (int r = 0; r < 8; ++r) {
            int m = tM * 64 + s * 16 + r + 8 * hi;   // D layout: m = r + 8*(lane>=16)
            float v = acc[s][r] + bi;
            if (act == 1) v = elu_f(v);
            if (scale)    v = v * bn_s + bn_t;
            if (residual) v += residual[(size_t)m * ldc + n];
            C[(size_t)m * ldc + n] = v;
        }
    }
}

// ---------------------------------------------------------------------------
// s_src / s_dst : per (row, head) 32-wide dot with a_src / a_dst
// ---------------------------------------------------------------------------
__global__ void srcdst_kernel(const float* __restrict__ Wh, const float* __restrict__ a_src,
                              const float* __restrict__ a_dst,
                              float* __restrict__ s_src, float* __restrict__ s_dst)
{
    int idx = blockIdx.x * blockDim.x + threadIdx.x;
    if (idx >= BN * NHEAD) return;
    int row = idx >> 2, h = idx & 3;
    const float* p = Wh + (size_t)row * DDIM + h * HDPER;
    float as = 0.f, ad = 0.f;
#pragma unroll
    for (int d2 = 0; d2 < HDPER; ++d2) { float v = p[d2]; as += v * a_src[d2]; ad += v * a_dst[d2]; }
    s_src[idx] = as; s_dst[idx] = ad;
}

// ---------------------------------------------------------------------------
// GAT attention: one block per (b,i).  Softmax over j (masked by adj),
// weighted aggregation of Wh, BN + ELU + residuals.
// ---------------------------------------------------------------------------
__global__ void gat_attn_kernel(const float* __restrict__ Wh, const float* __restrict__ s_src,
                                const float* __restrict__ s_dst, const float* __restrict__ adj,
                                const float* __restrict__ bn_g, const float* __restrict__ bn_b,
                                const float* __restrict__ extra_res, float* __restrict__ out)
{
    const int b = blockIdx.x / NNODE, i = blockIdx.x % NNODE;
    const int tid = threadIdx.x;                    // 128 threads
    __shared__ float alpha[NHEAD][128];
    __shared__ float red[128];

    const float* adjrow = adj + ((size_t)b * NNODE + i) * NNODE;
    const int j = tid;
    float aj   = (j < NNODE) ? adjrow[j] : 0.f;
    bool valid = (j < NNODE) && (fabsf(aj) >= 1e-6f);

    float ev[NHEAD];
#pragma unroll
    for (int h = 0; h < NHEAD; ++h) {
        float ssrc = s_src[((size_t)b * NNODE + i) * NHEAD + h];
        float sdst = (j < NNODE) ? s_dst[((size_t)b * NNODE + j) * NHEAD + h] : 0.f;
        float s  = ssrc + sdst;
        float lr = s > 0.f ? s : 0.2f * s;          // leaky_relu 0.2
        ev[h] = valid ? (lr + 0.5f * aj) : ((j < NNODE) ? -1e9f : -3.0e38f);
    }
#pragma unroll
    for (int h = 0; h < NHEAD; ++h) {
        red[tid] = ev[h]; __syncthreads();
        for (int s = 64; s > 0; s >>= 1) { if (tid < s) red[tid] = fmaxf(red[tid], red[tid + s]); __syncthreads(); }
        float mx = red[0]; __syncthreads();
        float ex = __expf(ev[h] - mx);
        red[tid] = ex; __syncthreads();
        for (int s = 64; s > 0; s >>= 1) { if (tid < s) red[tid] += red[tid + s]; __syncthreads(); }
        float sm = red[0]; __syncthreads();
        alpha[h][tid] = ex / sm;
    }
    __syncthreads();

    const int dd = tid, h2 = dd >> 5;
    const float* Whb = Wh + (size_t)b * NNODE * DDIM;
    float acc = 0.f;
    for (int jj = 0; jj < NNODE; ++jj) acc += alpha[h2][jj] * Whb[(size_t)jj * DDIM + dd];

    float v = acc * (bn_g[dd] * BN_RSQ) + bn_b[dd];
    v = elu_f(v);
    size_t o = ((size_t)b * NNODE + i) * DDIM + dd;
    v += Wh[o];
    if (extra_res) v += extra_res[o];
    out[o] = v;
}

// ---------------------------------------------------------------------------
__global__ void hmean_kernel(const float* __restrict__ h, float* __restrict__ hmean)
{
    int b = blockIdx.x, dd = threadIdx.x;           // 128 threads
    float acc = 0.f;
    for (int n = 0; n < NNODE; ++n) acc += h[((size_t)b * NNODE + n) * DDIM + dd];
    hmean[(size_t)b * DDIM + dd] = acc * (1.f / (float)NNODE);
}

__global__ void vn_init_kernel(const float* __restrict__ vnp, const float* __restrict__ hmean,
                               float* __restrict__ vn)
{
    int idx = blockIdx.x * blockDim.x + threadIdx.x;
    if (idx < BATCH * DDIM) vn[idx] = vnp[idx & (DDIM - 1)] + hmean[idx];
}

__global__ void vnu_kernel(const float* __restrict__ vnu_w, const float* __restrict__ vnu_b,
                           const float* __restrict__ hmean, float* __restrict__ vn)
{
    int b = blockIdx.x, tid = threadIdx.x;          // 128 threads
    __shared__ float z[2 * DDIM];
    z[tid]        = vn[(size_t)b * DDIM + tid];
    z[DDIM + tid] = hmean[(size_t)b * DDIM + tid];
    __syncthreads();
    float acc = vnu_b[tid];
    for (int k = 0; k < 2 * DDIM; ++k) acc += z[k] * vnu_w[(size_t)k * DDIM + tid];
    vn[(size_t)b * DDIM + tid] = elu_f(acc);
}

__global__ void addvn_kernel(float* __restrict__ h, const float* __restrict__ vn)
{
    size_t idx = (size_t)blockIdx.x * blockDim.x + threadIdx.x;
    if (idx >= (size_t)BN * DDIM) return;
    int dd = (int)(idx & (DDIM - 1));
    int b  = (int)(idx / ((size_t)NNODE * DDIM));
    h[idx] += 0.1f * vn[(size_t)b * DDIM + dd];
}

// ---------------------------------------------------------------------------
__global__ void pool_kernel(const float* __restrict__ h, const float* __restrict__ pm,
                            float* __restrict__ pooled)
{
    int b = blockIdx.x, dd = threadIdx.x;           // 128 threads
    float acc[KPOOL];
#pragma unroll
    for (int k = 0; k < KPOOL; ++k) acc[k] = 0.f;
    for (int n = 0; n < NNODE; ++n) {
        float hv = h[((size_t)b * NNODE + n) * DDIM + dd];
#pragma unroll
        for (int k = 0; k < KPOOL; ++k) acc[k] += hv * pm[n * KPOOL + k];
    }
#pragma unroll
    for (int k = 0; k < KPOOL; ++k) pooled[((size_t)b * KPOOL + k) * DDIM + dd] = acc[k];
}

__global__ void poolattn_kernel(const float* __restrict__ pooled, const float* __restrict__ w1,
                                const float* __restrict__ b1, const float* __restrict__ w2,
                                const float* __restrict__ b2, const float* __restrict__ vn,
                                float* __restrict__ flat)
{
    int b = blockIdx.x, tid = threadIdx.x;          // 128 threads
    __shared__ float pr[DDIM];
    __shared__ float logit[KPOOL];
    __shared__ float smx[KPOOL];
    for (int k = 0; k < KPOOL; ++k) {
        pr[tid] = pooled[((size_t)b * KPOOL + k) * DDIM + tid];
        __syncthreads();
        if (tid < 32) {
            float t = b1[tid];
            for (int d2 = 0; d2 < DDIM; ++d2) t += pr[d2] * w1[d2 * 32 + tid];
            float part = tanhf(t) * w2[tid];
#pragma unroll
            for (int off = 16; off > 0; off >>= 1) part += __shfl_down(part, off, 32);
            if (tid == 0) logit[k] = part + b2[0];
        }
        __syncthreads();
    }
    if (tid == 0) {
        float mx = logit[0];
        for (int k = 1; k < KPOOL; ++k) mx = fmaxf(mx, logit[k]);
        float s = 0.f;
        for (int k = 0; k < KPOOL; ++k) { smx[k] = __expf(logit[k] - mx); s += smx[k]; }
        for (int k = 0; k < KPOOL; ++k) smx[k] /= s;
    }
    __syncthreads();
    for (int k = 0; k < KPOOL; ++k)
        flat[(size_t)b * HEADDIM + k * DDIM + tid] = pooled[((size_t)b * KPOOL + k) * DDIM + tid] * smx[k];
    flat[(size_t)b * HEADDIM + KPOOL * DDIM + tid] = vn[(size_t)b * DDIM + tid];
}

__global__ void ln_kernel(const float* __restrict__ x, const float* __restrict__ g,
                          const float* __restrict__ bb, float* __restrict__ y)
{
    int b = blockIdx.x, tid = threadIdx.x;          // 256 threads
    __shared__ float rs[256], rq[256];
    float s = 0.f, q = 0.f;
    for (int i = tid; i < HEADDIM; i += 256) { float v = x[(size_t)b * HEADDIM + i]; s += v; q += v * v; }
    rs[tid] = s; rq[tid] = q; __syncthreads();
    for (int st = 128; st > 0; st >>= 1) { if (tid < st) { rs[tid] += rs[tid + st]; rq[tid] += rq[tid + st]; } __syncthreads(); }
    float mu   = rs[0] * (1.f / HEADDIM);
    float var  = rq[0] * (1.f / HEADDIM) - mu * mu;
    float rstd = rsqrtf(var + 1e-5f);
    for (int i = tid; i < HEADDIM; i += 256) {
        float v = x[(size_t)b * HEADDIM + i];
        y[(size_t)b * HEADDIM + i] = (v - mu) * rstd * g[i] + bb[i];
    }
}

__global__ void headout_kernel(const float* __restrict__ t2, const float* __restrict__ w,
                               const float* __restrict__ bb, float* __restrict__ out)
{
    int b = blockIdx.x, tid = threadIdx.x;          // 64 threads
    int col = tid & 1, p = tid >> 1;                // 32 partial groups per column
    float acc = 0.f;
    for (int k = p; k < 256; k += 32) acc += t2[(size_t)b * 256 + k] * w[k * 2 + col];
    __shared__ float r[64];
    r[tid] = acc; __syncthreads();
    for (int st = 16; st > 0; st >>= 1) { if (p < st) r[tid] += r[tid + 2 * st]; __syncthreads(); }
    if (p == 0) out[(size_t)b * 2 + col] = r[col] + bb[col];
}

// ---------------------------------------------------------------------------
static inline void launch_cvt_pad(hipStream_t stream, const float* src, _Float16* dst,
                                  int M, int srcK, int dstK)
{
    size_t total = (size_t)M * dstK;
    int blocks = (int)((total + 255) / 256);
    cvt_pad_kernel<<<blocks, 256, 0, stream>>>(src, dst, M, srcK, dstK);
}

static inline void launch_cvt_tr(hipStream_t stream, const float* src, _Float16* dst,
                                 int K, int Nn, int dstK)
{
    size_t total = (size_t)Nn * dstK;
    int blocks = (int)((total + 255) / 256);
    cvt_transpose_kernel<<<blocks, 256, 0, stream>>>(src, dst, K, Nn, dstK);
}

static inline void launch_gemm(hipStream_t stream, const _Float16* Ah, const _Float16* Bt, float* C,
                               int M, int Nn, int Kp, int ldc,
                               const float* bias, int act,
                               const float* scale, float smul, const float* shift,
                               const float* residual)
{
    int tiles  = (M / 64) * (Nn / 16);
    int blocks = (tiles + 3) / 4;                   // 4 waves (128 threads) per block
    gemm_wmma_kernel<<<blocks, 128, 0, stream>>>(Ah, Bt, C, M, Nn, Kp, ldc,
                                                 bias, act, scale, smul, shift, residual);
}

extern "C" void kernel_launch(void* const* d_in, const int* in_sizes, int n_in,
                              void* d_out, int out_size, void* d_ws, size_t ws_size,
                              hipStream_t stream)
{
    (void)in_sizes; (void)n_in; (void)out_size; (void)ws_size;

    // ---- inputs in setup_inputs() insertion order ----
    const float* x       = (const float*)d_in[0];   // [B,N,125]
    const float* adj     = (const float*)d_in[1];   // [B,N,N]
    const float* pm      = (const float*)d_in[2];   // [N,K]
    const float* enc_w   = (const float*)d_in[3];
    const float* enc_b   = (const float*)d_in[4];
    const float* bn_g    = (const float*)d_in[5];
    const float* bn_b    = (const float*)d_in[6];
    const float* vnp     = (const float*)d_in[7];
    // gat l (l=0..2): W = d_in[8+5l], a_src = 9+5l, a_dst = 10+5l, g = 11+5l, b = 12+5l
    const float* vnu_w   = (const float*)d_in[23];
    const float* vnu_b   = (const float*)d_in[24];
    const float* attn_w1 = (const float*)d_in[25];
    const float* attn_b1 = (const float*)d_in[26];
    const float* attn_w2 = (const float*)d_in[27];
    const float* attn_b2 = (const float*)d_in[28];
    const float* ln_g    = (const float*)d_in[29];
    const float* ln_b    = (const float*)d_in[30];
    // heads t=0..2: ad{w1,b1,w2,b2} = 31+8t..34+8t ; hd{w1,b1,w2,b2} = 35+8t..38+8t

    // ---- workspace carve-up ----
    char* wsb = (char*)d_ws;
    size_t off = 0;
    auto alloc = [&](size_t nbytes) -> void* {
        void* p = (void*)(wsb + off);
        off += (nbytes + 255) & ~(size_t)255;
        return p;
    };
    float* hb0       = (float*)alloc((size_t)BN * DDIM * 4);
    float* hb1       = (float*)alloc((size_t)BN * DDIM * 4);
    float* hb2       = (float*)alloc((size_t)BN * DDIM * 4);
    float* ssrc      = (float*)alloc((size_t)BN * NHEAD * 4);
    float* sdst      = (float*)alloc((size_t)BN * NHEAD * 4);
    float* hmean     = (float*)alloc((size_t)BATCH * DDIM * 4);
    float* vn        = (float*)alloc((size_t)BATCH * DDIM * 4);
    float* pooled    = (float*)alloc((size_t)BATCH * KPOOL * DDIM * 4);
    float* flatR     = (float*)alloc((size_t)BATCH * HEADDIM * 4);
    float* t1        = (float*)alloc((size_t)BATCH * 256 * 4);
    float* fa        = (float*)alloc((size_t)BATCH * HEADDIM * 4);
    float* t2        = (float*)alloc((size_t)BATCH * 256 * 4);
    _Float16* a16    = (_Float16*)alloc((size_t)BN * DDIM * 2);       // big A operand (f16)
    _Float16* flat16 = (_Float16*)alloc((size_t)BATCH * HEADDIM * 2); // layernormed flat (f16)
    _Float16* sm16   = (_Float16*)alloc((size_t)BATCH * HEADDIM * 2); // per-head A operand (f16)
    _Float16* b16    = (_Float16*)alloc((size_t)HEADDIM * 256 * 2);   // transposed weight (f16)

    // ---- encoder: h0 = bn(elu(x @ enc_w + enc_b)) ----
    launch_cvt_pad(stream, x, a16, BN, INDIM, DDIM);                  // pad 125 -> 128
    launch_cvt_tr(stream, enc_w, b16, INDIM, DDIM, DDIM);
    launch_gemm(stream, a16, b16, hb0, BN, DDIM, DDIM, DDIM,
                enc_b, /*act=*/1, bn_g, BN_RSQ, bn_b, nullptr);
    hmean_kernel<<<BATCH, DDIM, 0, stream>>>(hb0, hmean);
    vn_init_kernel<<<(BATCH * DDIM + 255) / 256, 256, 0, stream>>>(vnp, hmean, vn);

    // ---- 3 GAT layers with vn updates ----
    float* hcur = hb0; float* whb = hb1; float* hnext = hb2;
    for (int l = 0; l < 3; ++l) {
        const float* W  = (const float*)d_in[8  + 5 * l];
        const float* as = (const float*)d_in[9  + 5 * l];
        const float* ad = (const float*)d_in[10 + 5 * l];
        const float* gg = (const float*)d_in[11 + 5 * l];
        const float* gb = (const float*)d_in[12 + 5 * l];

        launch_cvt_pad(stream, hcur, a16, BN, DDIM, DDIM);
        launch_cvt_tr(stream, W, b16, DDIM, DDIM, DDIM);
        launch_gemm(stream, a16, b16, whb, BN, DDIM, DDIM, DDIM,
                    nullptr, 0, nullptr, 1.f, nullptr, nullptr);
        srcdst_kernel<<<(BN * NHEAD + 255) / 256, 256, 0, stream>>>(whb, as, ad, ssrc, sdst);
        gat_attn_kernel<<<BN, 128, 0, stream>>>(whb, ssrc, sdst, adj, gg, gb,
                                                (l == 0) ? nullptr : hcur, hnext);
        hmean_kernel<<<BATCH, DDIM, 0, stream>>>(hnext, hmean);
        vnu_kernel<<<BATCH, DDIM, 0, stream>>>(vnu_w, vnu_b, hmean, vn);
        if (l < 2) {
            int nb = (int)(((size_t)BN * DDIM + 255) / 256);
            addvn_kernel<<<nb, 256, 0, stream>>>(hnext, vn);
        }
        float* tmp = hcur; hcur = hnext; hnext = whb; whb = tmp;   // rotate buffers
    }
    // hcur == h3

    // ---- pooling + pooled attention + flat assembly + layernorm ----
    pool_kernel<<<BATCH, DDIM, 0, stream>>>(hcur, pm, pooled);
    poolattn_kernel<<<BATCH, DDIM, 0, stream>>>(pooled, attn_w1, attn_b1, attn_w2, attn_b2,
                                                vn, flatR);
    float* flatLN = (float*)d_out + 3 * BATCH * 2;   // output #4 slot doubles as GEMM input
    ln_kernel<<<BATCH, 256, 0, stream>>>(flatR, ln_g, ln_b, flatLN);
    launch_cvt_pad(stream, flatLN, flat16, BATCH, HEADDIM, HEADDIM);

    // ---- three classification heads ----
    for (int t = 0; t < 3; ++t) {
        const float* aw1  = (const float*)d_in[31 + 8 * t];
        const float* ab1  = (const float*)d_in[32 + 8 * t];
        const float* aw2  = (const float*)d_in[33 + 8 * t];
        const float* ab2  = (const float*)d_in[34 + 8 * t];
        const float* hw1  = (const float*)d_in[35 + 8 * t];
        const float* hb1p = (const float*)d_in[36 + 8 * t];
        const float* hw2  = (const float*)d_in[37 + 8 * t];
        const float* hb2p = (const float*)d_in[38 + 8 * t];

        // t1 = elu(flat @ aw1 + ab1)                  [512,1280]x[1280,256]
        launch_cvt_tr(stream, aw1, b16, HEADDIM, 256, HEADDIM);
        launch_gemm(stream, flat16, b16, t1, BATCH, 256, HEADDIM, 256,
                    ab1, 1, nullptr, 1.f, nullptr, nullptr);
        // fa = flat + t1 @ aw2 + ab2                  [512,256]x[256,1280]
        launch_cvt_pad(stream, t1, sm16, BATCH, 256, 256);
        launch_cvt_tr(stream, aw2, b16, 256, HEADDIM, 256);
        launch_gemm(stream, sm16, b16, fa, BATCH, HEADDIM, 256, HEADDIM,
                    ab2, 0, nullptr, 1.f, nullptr, flatLN);
        // t2 = elu(fa @ hw1 + hb1)                    [512,1280]x[1280,256]
        launch_cvt_pad(stream, fa, sm16, BATCH, HEADDIM, HEADDIM);
        launch_cvt_tr(stream, hw1, b16, HEADDIM, 256, HEADDIM);
        launch_gemm(stream, sm16, b16, t2, BATCH, 256, HEADDIM, 256,
                    hb1p, 1, nullptr, 1.f, nullptr, nullptr);
        // out = t2 @ hw2 + hb2                        [512,256]x[256,2]
        headout_kernel<<<BATCH, 64, 0, stream>>>(t2, hw2, hb2p,
                                                 (float*)d_out + (size_t)t * BATCH * 2);
    }
}
